// AttentionBlock_12060268167256
// MI455X (gfx1250) — matrix operand order repaired and checked by reading, compiled, not run on hardware
//
#include <hip/hip_runtime.h>

// ---------------------------------------------------------------------------
// CDNA5 (gfx1250) attention block: GroupNorm -> QKV GEMM -> flash attention
// -> proj GEMM (+bias +residual).  All matmuls via v_wmma_f32_16x16x32_bf16.
// wave32; C/D frag: lane(lo,hi), c[r] = D[M = r+8*hi][N = lo]
// A frag (16x32 bf16): a[j] = A[M = lo][K = j + 16*hi]
// B frag (32x16 bf16): b[j] = B[K = j + 16*hi][N = lo]
// ---------------------------------------------------------------------------

typedef __attribute__((ext_vector_type(16))) __bf16 v16bf;
typedef __attribute__((ext_vector_type(8)))  __bf16 v8bf;
typedef __attribute__((ext_vector_type(4)))  __bf16 v4bf;
typedef __attribute__((ext_vector_type(2)))  __bf16 v2bf;
typedef __attribute__((ext_vector_type(8)))  float  v8f;

#define B_DIM   8
#define C_DIM   512
#define N_DIM   1024          // H*W
#define NH      8
#define HD      64
#define GROUPS  32
#define GSIZE   16            // channels per group
#define LOG2E   1.44269504088896340736f

// Async global->LDS copy (GLOBAL_LOAD_ASYNC_TO_LDS_B128, ASYNCcnt-tracked).
#define USE_ASYNC_COPY 1

__device__ __forceinline__ v16bf cat8(v8bf a, v8bf b) {
  return __builtin_shufflevector(a, b, 0,1,2,3,4,5,6,7,8,9,10,11,12,13,14,15);
}

__device__ __forceinline__ v8f wmma_bf16(v16bf a, v16bf b, v8f c) {
  // (neg_a, A, neg_b, B, c_mod, C, reuse_a, reuse_b)
  return __builtin_amdgcn_wmma_f32_16x16x32_bf16(false, a, false, b, (short)0, c,
                                                 false, false);
}

__device__ __forceinline__ void async_copy_b128(const __bf16* gsrc, __bf16* lds_dst) {
#if USE_ASYNC_COPY
  unsigned lds_off =
      (unsigned)(size_t)(__attribute__((address_space(3))) __bf16*)lds_dst;
  asm volatile("global_load_async_to_lds_b128 %0, %1, off"
               :: "v"(lds_off), "v"(gsrc) : "memory");
#else
  *(v8bf*)lds_dst = *(const v8bf*)gsrc;
#endif
}

__device__ __forceinline__ void async_copy_fence() {
#if USE_ASYNC_COPY
  asm volatile("s_wait_asynccnt 0x0" ::: "memory");
#endif
}

// ---------------------------------------------------------------------------
// Kernel 1: convert fp32 weights to bf16 (qkv_w 1536x512, proj_w 512x512)
// ---------------------------------------------------------------------------
#define QW_N (3 * C_DIM * C_DIM)   // 786432
#define PW_N (C_DIM * C_DIM)       // 262144

__global__ void __launch_bounds__(256)
convw_kernel(const float* __restrict__ qw, const float* __restrict__ pw,
             __bf16* __restrict__ qwb, __bf16* __restrict__ pwb) {
  const int n4q = QW_N / 4, n4p = PW_N / 4;
  int i = blockIdx.x * blockDim.x + threadIdx.x;
  const int stride = gridDim.x * blockDim.x;
  for (; i < n4q + n4p; i += stride) {
    const float4* src; __bf16* dst; int e;
    if (i < n4q) { src = (const float4*)qw; dst = qwb; e = i; }
    else         { src = (const float4*)pw; dst = pwb; e = i - n4q; }
    float4 v = src[e];
    v4bf o; o[0] = (__bf16)v.x; o[1] = (__bf16)v.y;
            o[2] = (__bf16)v.z; o[3] = (__bf16)v.w;
    *(v4bf*)(dst + (size_t)e * 4) = o;
  }
}

// ---------------------------------------------------------------------------
// Kernel 2: fused GroupNorm.  One block per (batch, group): reduce over
// 16 channels x 1024 spatial = 16384 floats, then normalize -> bf16 xn[b][c][n]
// ---------------------------------------------------------------------------
__global__ void __launch_bounds__(256)
gn_kernel(const float* __restrict__ x, const float* __restrict__ w,
          const float* __restrict__ bgn, __bf16* __restrict__ xn) {
  const int b = blockIdx.x >> 5;       // /GROUPS
  const int g = blockIdx.x & 31;
  const size_t base = ((size_t)b * C_DIM + g * GSIZE) * N_DIM;
  const int tid = threadIdx.x;
  const float4* x4 = (const float4*)(x + base);

  float s = 0.f, sq = 0.f;
  for (int e = tid; e < (GSIZE * N_DIM) / 4; e += 256) {
    float4 v = x4[e];
    s  += v.x + v.y + v.z + v.w;
    sq += v.x * v.x + v.y * v.y + v.z * v.z + v.w * v.w;
  }
#pragma unroll
  for (int o = 16; o >= 1; o >>= 1) {
    s  += __shfl_xor(s,  o, 32);
    sq += __shfl_xor(sq, o, 32);
  }
  __shared__ float ss[8], ssq[8];
  const int wave = tid >> 5;
  if ((tid & 31) == 0) { ss[wave] = s; ssq[wave] = sq; }
  __syncthreads();
  float tot = 0.f, totq = 0.f;
#pragma unroll
  for (int i = 0; i < 8; ++i) { tot += ss[i]; totq += ssq[i]; }
  const float mean = tot * (1.f / 16384.f);
  const float var  = totq * (1.f / 16384.f) - mean * mean;
  const float rstd = rsqrtf(var + 1e-5f);

  for (int e = tid; e < (GSIZE * N_DIM) / 4; e += 256) {
    float4 v = x4[e];
    const int cg = g * GSIZE + (e >> 8);      // (e*4) >> 10
    const float sc = rstd * w[cg];
    const float sb = bgn[cg] - mean * sc;
    v4bf o;
    o[0] = (__bf16)(v.x * sc + sb); o[1] = (__bf16)(v.y * sc + sb);
    o[2] = (__bf16)(v.z * sc + sb); o[3] = (__bf16)(v.w * sc + sb);
    *(v4bf*)(xn + base + (size_t)e * 4) = o;
  }
}

// ---------------------------------------------------------------------------
// Shared WMMA GEMM core: Y[o][n] = sum_c W[o][c] * X[c][n], K = C_DIM = 512.
// Block = 256 threads = 8 waves computes a 64(o) x 128(n) tile.
// A-frags read directly from global (per-lane contiguous W rows).
// X staged TRANSPOSED in ping-pong LDS; each thread loads two adjacent k-rows
// and stores packed k-pairs (ds_store_b32), so B-frags are single 32B reads.
// One block barrier per 32-wide K-step; global loads overlap WMMA compute.
// ---------------------------------------------------------------------------
#define XT_STRIDE (128 * 32)

__device__ __forceinline__ void
gemm_tile_core(const __bf16* __restrict__ W, const __bf16* __restrict__ X,
               int o0, int n0, v8f (&acc)[4], __bf16* __restrict__ Xt) {
  const int tid  = threadIdx.x;
  const int lane = tid & 31;
  const int lo = lane & 15, hi = lane >> 4;
  const int wave = tid >> 5;
  const int ow = (wave & 3) * 16;
  const int nw = (wave >> 2) * 64;
  const int kp  = (tid >> 4) << 1;   // 0..30, even (k-row pair)
  const int nc8 = (tid & 15) * 8;    // 0..120

  const __bf16* Wrow = W + (size_t)(o0 + ow + lo) * C_DIM;

  // prologue: stage step 0
  v8bf xa = *(const v8bf*)(X + (size_t)kp * N_DIM + n0 + nc8);
  v8bf xb = *(const v8bf*)(X + (size_t)(kp + 1) * N_DIM + n0 + nc8);
#pragma unroll
  for (int i = 0; i < 8; ++i) {
    v2bf p; p[0] = xa[i]; p[1] = xb[i];
    *(v2bf*)(Xt + (nc8 + i) * 32 + kp) = p;
  }
  __syncthreads();

#pragma unroll 2
  for (int s = 0; s < 16; ++s) {
    const int c0 = s * 32;
    if (s + 1 < 16) {   // prefetch next X slice into registers
      xa = *(const v8bf*)(X + (size_t)(c0 + 32 + kp) * N_DIM + n0 + nc8);
      xb = *(const v8bf*)(X + (size_t)(c0 + 33 + kp) * N_DIM + n0 + nc8);
    }

    __bf16* buf = Xt + (s & 1) * XT_STRIDE;
    v16bf a = cat8(*(const v8bf*)(Wrow + c0 + 16 * hi),
                   *(const v8bf*)(Wrow + c0 + 8 + 16 * hi));
    v16bf bf[4];
#pragma unroll
    for (int t = 0; t < 4; ++t)
      bf[t] = *(const v16bf*)(buf + (nw + t * 16 + lo) * 32 + 16 * hi);
#pragma unroll
    for (int t = 0; t < 4; ++t) acc[t] = wmma_bf16(a, bf[t], acc[t]);

    if (s + 1 < 16) {
      __bf16* nbuf = Xt + ((s + 1) & 1) * XT_STRIDE;
#pragma unroll
      for (int i = 0; i < 8; ++i) {
        v2bf p; p[0] = xa[i]; p[1] = xb[i];
        *(v2bf*)(nbuf + (nc8 + i) * 32 + kp) = p;
      }
    }
    __syncthreads();
  }
}

// ---------------------------------------------------------------------------
// Kernel 3: QKV GEMM.  Output into q/k/v buffers, layout [b*NH+h][n][d] bf16.
// q pre-scaled by hd^-0.5 * log2(e) (softmax done base-2; v_exp_f32 is exp2).
// Epilogue: wave-private LDS micro-transpose -> coalesced 16B stores.
// Grid: (1536/64) * (1024/128) * B = 1536 blocks.
// ---------------------------------------------------------------------------
__global__ void __launch_bounds__(256)
qkv_gemm_kernel(const __bf16* __restrict__ W, const __bf16* __restrict__ Xall,
                const float* __restrict__ bias,
                __bf16* __restrict__ qb, __bf16* __restrict__ kb,
                __bf16* __restrict__ vb) {
  __shared__ __align__(32) __bf16 Xt[2 * XT_STRIDE];
  __shared__ __align__(32) __bf16 Ot[8][16 * 16];   // per-wave C-tile transpose

  const int idx = blockIdx.x;
  const int b   = idx / 192;
  const int rem = idx % 192;
  const int o0  = (rem >> 3) * 64;
  const int n0  = (rem & 7) * 128;
  const __bf16* X = Xall + (size_t)b * C_DIM * N_DIM;

  v8f acc[4];
#pragma unroll
  for (int t = 0; t < 4; ++t) { v8f z = {}; acc[t] = z; }
  gemm_tile_core(W, X, o0, n0, acc, Xt);

  const int lane = threadIdx.x & 31, wave = threadIdx.x >> 5;
  const int lo = lane & 15, hi = lane >> 4;
  const int ow = (wave & 3) * 16, nw = (wave >> 2) * 64;

  const int obase = o0 + ow;                  // 16-aligned: uniform head/which
  const int which = obase >> 9;               // 0=q,1=k,2=v
  const int hh    = (obase & 511) >> 6;
  const int dbase = obase & 63;
  const float qscale = (which == 0) ? 0.125f * LOG2E : 1.0f;
  __bf16* dst = (which == 0) ? qb : (which == 1) ? kb : vb;

  float bv8[8];
#pragma unroll
  for (int r = 0; r < 8; ++r) bv8[r] = bias[obase + r + 8 * hi];

#pragma unroll
  for (int t = 0; t < 4; ++t) {
    // stage tile transposed: Ot[n][o]
#pragma unroll
    for (int r = 0; r < 8; ++r)
      Ot[wave][lo * 16 + r + 8 * hi] = (__bf16)((acc[t][r] + bv8[r]) * qscale);
    __builtin_amdgcn_wave_barrier();
    v8bf ov = *(const v8bf*)(&Ot[wave][(lane & 15) * 16 + (lane >> 4) * 8]);
    __builtin_amdgcn_wave_barrier();
    const int n = n0 + nw + t * 16 + (lane & 15);
    const int d = dbase + (lane >> 4) * 8;
    *(v8bf*)(dst + (((size_t)b * NH + hh) * N_DIM + n) * HD + d) = ov;
  }
}

// ---------------------------------------------------------------------------
// Kernel 4: flash attention.  Block = 8 waves; each wave owns TWO 16-query
// tiles (32 queries) of a 256-query block for one (b,h).  Per 32-key step:
//   - K tile (32x64) copied with GLOBAL_LOAD_ASYNC_TO_LDS_B128 into ping-pong
//     LDS (ASYNCcnt fenced before the block barrier); V tile transposed
//     (Vt[d][key]) with packed ds_store_b32,
//   - 8 WMMAs for S (K/V frag reads amortized over both q-tiles),
//   - online softmax in fp32 (exp2-domain, 16-lane shfl),
//   - P -> bf16 A-frags via wave-private LDS (DS in-order within a wave),
//   - 8 WMMAs for O.  One block barrier per step.
// Epilogue: wave-private LDS transpose -> coalesced 16B stores to ao[b][c][n].
// Grid: B*NH*(1024/256) = 256 blocks (halves redundant K/V HBM traffic).
// ---------------------------------------------------------------------------
__global__ void __launch_bounds__(256)
attn_kernel(const __bf16* __restrict__ qbuf, const __bf16* __restrict__ kbuf,
            const __bf16* __restrict__ vbuf, __bf16* __restrict__ ao) {
  __shared__ __align__(32) __bf16 Kt[2][32 * 64];     // [key][d] direct copy
  __shared__ __align__(32) __bf16 Vt[2][64 * 32];     // [d][key] transposed
  __shared__ __align__(32) __bf16 Pt[8][2 * 16 * 32]; // per-wave P tiles [u][q][key]

  const int tid = threadIdx.x;
  const int lane = tid & 31, wave = tid >> 5;
  const int lo = lane & 15, hi = lane >> 4;

  const int idx = blockIdx.x;
  const int qt = idx & 3;
  const int bh = idx >> 2;             // b*NH + h
  const int h = bh & 7, b = bh >> 3;
  const size_t base = (size_t)bh * N_DIM * HD;
  const int q0 = qt * 256 + wave * 32;

  // Q A-frags for both 16-row tiles (held for whole loop)
  v16bf aq[2][2];
#pragma unroll
  for (int u = 0; u < 2; ++u) {
    const __bf16* Qr = qbuf + base + (size_t)(q0 + u * 16 + lo) * HD;
    aq[u][0] = cat8(*(const v8bf*)(Qr + 16 * hi),
                    *(const v8bf*)(Qr + 8 + 16 * hi));
    aq[u][1] = cat8(*(const v8bf*)(Qr + 32 + 16 * hi),
                    *(const v8bf*)(Qr + 40 + 16 * hi));
  }

  v8f acc[2][4];
#pragma unroll
  for (int u = 0; u < 2; ++u)
#pragma unroll
    for (int t = 0; t < 4; ++t) { v8f z = {}; acc[u][t] = z; }
  float m[16], l[16];
#pragma unroll
  for (int r = 0; r < 16; ++r) { m[r] = -3.0e38f; l[r] = 0.f; }

  // staging coords: K copy = 1 key row x 8 d per thread;
  //                 V transpose = 2 adjacent key rows x 4 d per thread
  const int skey = tid >> 3;           // 0..31
  const int sd   = (tid & 7) * 8;      // 0..56
  const int kp2  = (tid & 15) * 2;     // 0..30, even
  const int dg   = (tid >> 4) * 4;     // 0..60

  // prologue: stage step 0
  async_copy_b128(kbuf + base + (size_t)skey * HD + sd, &Kt[0][skey * 64 + sd]);
  {
    const __bf16* vrow = vbuf + base + (size_t)kp2 * HD + dg;
    v4bf va  = *(const v4bf*)(vrow);
    v4bf vb2 = *(const v4bf*)(vrow + HD);
#pragma unroll
    for (int i = 0; i < 4; ++i) {
      v2bf p; p[0] = va[i]; p[1] = vb2[i];
      *(v2bf*)(&Vt[0][(dg + i) * 32 + kp2]) = p;
    }
  }
  async_copy_fence();
  __syncthreads();

  v4bf va = {}, vb2 = {};
#pragma unroll 2
  for (int it = 0; it < N_DIM / 32; ++it) {
    const int cur = it & 1;
    if (it + 1 < N_DIM / 32) {   // prefetch next tile (K async, V to regs)
      async_copy_b128(kbuf + base + (size_t)((it + 1) * 32 + skey) * HD + sd,
                      &Kt[cur ^ 1][skey * 64 + sd]);
      const __bf16* vrow = vbuf + base + (size_t)((it + 1) * 32 + kp2) * HD + dg;
      va  = *(const v4bf*)(vrow);
      vb2 = *(const v4bf*)(vrow + HD);
    }

    // S = Q K^T for 32 keys (two 16-key column tiles, d split 2 x K=32)
    const __bf16* kb_ = &Kt[cur][0];
    v16bf bk00 = *(const v16bf*)(kb_ + lo * 64 + 16 * hi);
    v16bf bk01 = *(const v16bf*)(kb_ + lo * 64 + 32 + 16 * hi);
    v16bf bk10 = *(const v16bf*)(kb_ + (16 + lo) * 64 + 16 * hi);
    v16bf bk11 = *(const v16bf*)(kb_ + (16 + lo) * 64 + 32 + 16 * hi);
    v8f s0[2], s1[2];
#pragma unroll
    for (int u = 0; u < 2; ++u) {
      v8f z = {};
      s0[u] = wmma_bf16(aq[u][0], bk00, z);
      s0[u] = wmma_bf16(aq[u][1], bk01, s0[u]);
      s1[u] = wmma_bf16(aq[u][0], bk10, z);
      s1[u] = wmma_bf16(aq[u][1], bk11, s1[u]);
    }

    // online softmax (base-2 domain; q was pre-scaled by 0.125*log2e)
#pragma unroll
    for (int u = 0; u < 2; ++u) {
#pragma unroll
      for (int r = 0; r < 8; ++r) {
        float v0 = s0[u][r], v1 = s1[u][r];
        float mx = fmaxf(v0, v1);
#pragma unroll
        for (int o = 8; o >= 1; o >>= 1) mx = fmaxf(mx, __shfl_xor(mx, o, 32));
        const int  ur = u * 8 + r;
        const float mn = fmaxf(m[ur], mx);
        const float al = __builtin_amdgcn_exp2f(m[ur] - mn);
        const float p0 = __builtin_amdgcn_exp2f(v0 - mn);
        const float p1 = __builtin_amdgcn_exp2f(v1 - mn);
        float ps = p0 + p1;
#pragma unroll
        for (int o = 8; o >= 1; o >>= 1) ps += __shfl_xor(ps, o, 32);
        l[ur] = l[ur] * al + ps;
        m[ur] = mn;
        acc[u][0][r] *= al; acc[u][1][r] *= al;
        acc[u][2][r] *= al; acc[u][3][r] *= al;
        Pt[wave][u * 512 + (r + 8 * hi) * 32 + lo]      = (__bf16)p0;
        Pt[wave][u * 512 + (r + 8 * hi) * 32 + 16 + lo] = (__bf16)p1;
      }
    }
    __builtin_amdgcn_wave_barrier();   // wave-private LDS: order store->load

    // P in A-frag layout: a[j] = P[q=lo][key=j+16*hi]
    v16bf ap[2];
#pragma unroll
    for (int u = 0; u < 2; ++u)
      ap[u] = cat8(*(const v8bf*)(&Pt[wave][u * 512 + lo * 32 + 16 * hi]),
                   *(const v8bf*)(&Pt[wave][u * 512 + lo * 32 + 8 + 16 * hi]));
    __builtin_amdgcn_wave_barrier();   // order load -> next iter's store

    // O += P(16x32) x V(32x64); B-frag = single 32B read from Vt[d][key]
    v16bf bv[4];
#pragma unroll
    for (int t = 0; t < 4; ++t)
      bv[t] = *(const v16bf*)(&Vt[cur][(t * 16 + lo) * 32 + 16 * hi]);
#pragma unroll
    for (int u = 0; u < 2; ++u)
#pragma unroll
      for (int t = 0; t < 4; ++t) acc[u][t] = wmma_bf16(ap[u], bv[t], acc[u][t]);

    if (it + 1 < N_DIM / 32) {   // store V regs into the other buffer
      const int nxt = cur ^ 1;
#pragma unroll
      for (int i = 0; i < 4; ++i) {
        v2bf p; p[0] = va[i]; p[1] = vb2[i];
        *(v2bf*)(&Vt[nxt][(dg + i) * 32 + kp2]) = p;
      }
    }
    async_copy_fence();
    __syncthreads();
  }

  float inv[16];
#pragma unroll
  for (int r = 0; r < 16; ++r) inv[r] = 1.0f / l[r];

  // epilogue: per-tile LDS transpose (reuse Pt) -> coalesced 16B stores
#pragma unroll
  for (int u = 0; u < 2; ++u) {
#pragma unroll
    for (int t = 0; t < 4; ++t) {
#pragma unroll
      for (int r = 0; r < 8; ++r)
        Pt[wave][lo * 16 + r + 8 * hi] = (__bf16)(acc[u][t][r] * inv[u * 8 + r]);
      __builtin_amdgcn_wave_barrier();
      v8bf ov = *(const v8bf*)(&Pt[wave][(lane & 15) * 16 + (lane >> 4) * 8]);
      __builtin_amdgcn_wave_barrier();
      const int c = h * HD + t * 16 + (lane & 15);
      *(v8bf*)(ao + ((size_t)b * C_DIM + c) * N_DIM +
               q0 + u * 16 + (lane >> 4) * 8) = ov;
    }
  }
}

// ---------------------------------------------------------------------------
// Kernel 5: proj GEMM + bias + residual -> fp32 output [b][c][n].
// Grid: (512/64)*(1024/128)*B = 512 blocks.
// ---------------------------------------------------------------------------
__global__ void __launch_bounds__(256)
proj_gemm_kernel(const __bf16* __restrict__ W, const __bf16* __restrict__ Xall,
                 const float* __restrict__ bias, const float* __restrict__ resid,
                 float* __restrict__ out) {
  __shared__ __align__(32) __bf16 Xt[2 * XT_STRIDE];

  const int idx = blockIdx.x;
  const int b   = idx / 64;
  const int rem = idx % 64;
  const int o0  = (rem >> 3) * 64;
  const int n0  = (rem & 7) * 128;
  const __bf16* X = Xall + (size_t)b * C_DIM * N_DIM;

  v8f acc[4];
#pragma unroll
  for (int t = 0; t < 4; ++t) { v8f z = {}; acc[t] = z; }
  gemm_tile_core(W, X, o0, n0, acc, Xt);

  const int lane = threadIdx.x & 31, wave = threadIdx.x >> 5;
  const int lo = lane & 15, hi = lane >> 4;
  const int ow = (wave & 3) * 16, nw = (wave >> 2) * 64;
#pragma unroll
  for (int t = 0; t < 4; ++t) {
    const int n = n0 + nw + t * 16 + lo;
#pragma unroll
    for (int r = 0; r < 8; ++r) {
      const int o = o0 + ow + r + 8 * hi;
      const size_t di = ((size_t)b * C_DIM + o) * N_DIM + n;
      out[di] = acc[t][r] + bias[o] + resid[di];
    }
  }
}

// ---------------------------------------------------------------------------
// Host launcher
// ---------------------------------------------------------------------------
extern "C" void kernel_launch(void* const* d_in, const int* in_sizes, int n_in,
                              void* d_out, int out_size, void* d_ws, size_t ws_size,
                              hipStream_t stream) {
  const float* x      = (const float*)d_in[0];
  const float* gn_w   = (const float*)d_in[1];
  const float* gn_b   = (const float*)d_in[2];
  const float* qkv_w  = (const float*)d_in[3];
  const float* qkv_b  = (const float*)d_in[4];
  const float* proj_w = (const float*)d_in[5];
  const float* proj_b = (const float*)d_in[6];
  float* out = (float*)d_out;

  char* ws = (char*)d_ws;
  size_t off = 0;
  auto take = [&](size_t bytes) {
    char* p = ws + off;
    off = (off + bytes + 255) & ~(size_t)255;
    return p;
  };
  const size_t XN_B = (size_t)B_DIM * C_DIM * N_DIM * 2;        // 8 MB bf16
  const size_t HB_B = (size_t)B_DIM * NH * N_DIM * HD * 2;      // 8 MB bf16
  __bf16* xn  = (__bf16*)take(XN_B);
  __bf16* qwb = (__bf16*)take((size_t)QW_N * 2);
  __bf16* pwb = (__bf16*)take((size_t)PW_N * 2);
  __bf16* qb  = (__bf16*)take(HB_B);
  __bf16* kb  = (__bf16*)take(HB_B);
  __bf16* vb  = (__bf16*)take(HB_B);
  __bf16* ao  = (__bf16*)take(XN_B);
  (void)ws_size; (void)in_sizes; (void)n_in; (void)out_size;

  convw_kernel<<<512, 256, 0, stream>>>(qkv_w, proj_w, qwb, pwb);
  gn_kernel<<<B_DIM * GROUPS, 256, 0, stream>>>(x, gn_w, gn_b, xn);
  qkv_gemm_kernel<<<24 * 8 * B_DIM, 256, 0, stream>>>(qwb, xn, qkv_b, qb, kb, vb);
  attn_kernel<<<B_DIM * NH * (N_DIM / 256), 256, 0, stream>>>(qb, kb, vb, ao);
  proj_gemm_kernel<<<8 * 8 * B_DIM, 256, 0, stream>>>(pwb, ao, proj_b, x, out);
}
